// DCGAN_G_74801150427212
// MI455X (gfx1250) — matrix operand order repaired and checked by reading, not compiled, not run on hardware
//
#include <hip/hip_runtime.h>
#include <hip/hip_bf16.h>
#include <math.h>

typedef _Float16 v16h __attribute__((ext_vector_type(16)));
typedef _Float16 v8h  __attribute__((ext_vector_type(8)));
typedef float    v8f  __attribute__((ext_vector_type(8)));

#define WAVE 32
#define NN   69      // neurons
#define EPS  1e-5f

// ---------------------------------------------------------------------------
// Phase 1/3: memory-bound GEMV (one wave per row, coalesced row reads)
// ---------------------------------------------------------------------------
__global__ void gemv_kernel(const float* __restrict__ W, const float* __restrict__ x,
                            const float* __restrict__ bias, float* __restrict__ out,
                            int rows, int K) {
  int wid  = (blockIdx.x * blockDim.x + threadIdx.x) / WAVE;
  int lane = threadIdx.x & (WAVE - 1);
  if (wid >= rows) return;
  const float* Wr = W + (size_t)wid * (size_t)K;
  float acc = 0.f;
  for (int k = lane; k < K; k += WAVE) acc += Wr[k] * x[k];
  #pragma unroll
  for (int off = WAVE / 2; off > 0; off >>= 1) acc += __shfl_down(acc, off, WAVE);
  if (lane == 0) out[wid] = acc + bias[wid];
}

// ---------------------------------------------------------------------------
// Phase 2: nearest-up2 fused with 4x4 conv, asymmetric SAME pad (1,2)
// out[o][y][x] = sum_{c,ky,kx} in[c][(y-1+ky)/2][(x-1+kx)/2] * w[o][c][ky][kx]
// ---------------------------------------------------------------------------
__global__ void upconv_kernel(const float* __restrict__ in, const float* __restrict__ w,
                              float* __restrict__ out, int Cin, int Cout, int S) {
  int t = blockIdx.x * blockDim.x + threadIdx.x;
  int total = Cout * S * S;
  if (t >= total) return;
  int x = t % S, y = (t / S) % S, o = t / (S * S);
  int inS = S >> 1;
  float acc = 0.f;
  for (int c = 0; c < Cin; ++c) {
    const float* ic = in + (size_t)c * inS * inS;
    const float* wc = w + ((size_t)o * Cin + c) * 16;
    #pragma unroll
    for (int ky = 0; ky < 4; ++ky) {
      int yy = y - 1 + ky;
      if (yy < 0 || yy >= S) continue;
      const float* ir = ic + (size_t)(yy >> 1) * inS;
      #pragma unroll
      for (int kx = 0; kx < 4; ++kx) {
        int xx = x - 1 + kx;
        if (xx < 0 || xx >= S) continue;
        acc += ir[xx >> 1] * wc[ky * 4 + kx];
      }
    }
  }
  out[t] = acc;
}

// ---------------------------------------------------------------------------
// BatchNorm training-mode stats (biased var), one block per channel
// ---------------------------------------------------------------------------
__global__ void bnstats_kernel(const float* __restrict__ buf, float* __restrict__ mean,
                               float* __restrict__ rstd, int HW) {
  int c = blockIdx.x;
  const float* p = buf + (size_t)c * HW;
  float s = 0.f, sq = 0.f;
  for (int i = threadIdx.x; i < HW; i += blockDim.x) { float v = p[i]; s += v; sq += v * v; }
  __shared__ float ss[256], ssq[256];
  ss[threadIdx.x] = s; ssq[threadIdx.x] = sq;
  __syncthreads();
  for (int o = 128; o > 0; o >>= 1) {
    if ((int)threadIdx.x < o) { ss[threadIdx.x] += ss[threadIdx.x + o];
                                ssq[threadIdx.x] += ssq[threadIdx.x + o]; }
    __syncthreads();
  }
  if (threadIdx.x == 0) {
    float m = ss[0] / (float)HW;
    float v = ssq[0] / (float)HW - m * m;
    mean[c] = m;
    rstd[c] = rsqrtf(v + EPS);
  }
}

__global__ void bnrelu_kernel(float* __restrict__ buf, const float* __restrict__ mean,
                              const float* __restrict__ rstd, const float* __restrict__ g,
                              const float* __restrict__ b, int HW, int total) {
  int t = blockIdx.x * blockDim.x + threadIdx.x;
  if (t >= total) return;
  int c = t / HW;
  float v = (buf[t] - mean[c]) * rstd[c] * g[c] + b[c];
  buf[t] = fmaxf(v, 0.f);
}

// ---------------------------------------------------------------------------
// Final 4x4 conv (no upsample), 1 output channel, tanh
// ---------------------------------------------------------------------------
__global__ void conv_tanh_kernel(const float* __restrict__ in, const float* __restrict__ w,
                                 float* __restrict__ out, int Cin, int S) {
  int t = blockIdx.x * blockDim.x + threadIdx.x;
  if (t >= S * S) return;
  int x = t % S, y = t / S;
  float acc = 0.f;
  for (int c = 0; c < Cin; ++c) {
    const float* ic = in + (size_t)c * S * S;
    const float* wc = w + (size_t)c * 16;
    #pragma unroll
    for (int ky = 0; ky < 4; ++ky) {
      int yy = y - 1 + ky;
      if (yy < 0 || yy >= S) continue;
      #pragma unroll
      for (int kx = 0; kx < 4; ++kx) {
        int xx = x - 1 + kx;
        if (xx < 0 || xx >= S) continue;
        acc += ic[yy * S + xx] * wc[ky * 4 + kx];
      }
    }
  }
  out[t] = tanhf(acc);
}

// ---------------------------------------------------------------------------
// Sequential spiking recurrence: single wave, WMMA matvec per step.
//   new = -prev + tanh(prev @ w),  w: 69x69, 99800 steps.
// B (=w in f16) padded to 96x80 lives in 15 resident v16h tiles.
// A is rebuilt each step from the 96-half state vector in LDS; every lane
// fills its A row with the same prev vector, so every D row is valid and
// we read M=0 (v8f element 0, lanes 0-15).
// ---------------------------------------------------------------------------
__device__ __forceinline__ v8f wmma_f16(v16h a, v16h b, v8f c) {
  return __builtin_amdgcn_wmma_f32_16x16x32_f16(false, a, false, b, (short)0, c,
                                                false, false);
}

// Branch-free tanh on the serial critical path:
//   tanh(x) = 1 - 2/(e^{2x}+1),  clamp so exp never overflows.
// med3 + mul + v_exp_f32 + add + v_rcp_f32 + fma + sub: no IEEE div sequence,
// no exec-mask divergence.
__device__ __forceinline__ float tanh_fast(float x) {
  float xc = fminf(fmaxf(x, -9.0f), 9.0f);
  float e  = __expf(2.0f * xc);
  return __builtin_fmaf(-2.0f, __builtin_amdgcn_rcpf(e + 1.0f), 1.0f);
}

// Intra-wave LDS ordering without a workgroup barrier: LDS ops from one wave
// execute in order in hardware, so we only need to pin compiler code motion.
__device__ __forceinline__ void wave_lds_fence() {
  __builtin_amdgcn_fence(__ATOMIC_ACQ_REL, "wavefront");
  __builtin_amdgcn_wave_barrier();
}

// B tile 32x16 in this code's contiguous k order (A matches): lanes 0-15 hold K=0..15 (half e -> K=e),
// lanes 16-31 hold K=16..31; lane%16 = N.
__device__ __forceinline__ v16h load_btile(const float* __restrict__ w69, int lane,
                                           int kt, int nt) {
  int kbase = 32 * kt + ((lane & 16) ? 16 : 0);
  int n     = 16 * nt + (lane & 15);
  v16h r;
  #pragma unroll
  for (int e = 0; e < 16; ++e) {
    int K = kbase + e;
    float v = (K < NN && n < NN) ? w69[K * NN + n] : 0.0f;
    r[e] = (_Float16)v;
  }
  return r;
}

// A tile 16x32 in this code's contiguous k order (B matches): lane<16 holds K=0..15, lane>=16 holds
// K=16..31; two aligned 8-half LDS segments per tile.
__device__ __forceinline__ v16h load_atile(const _Float16* __restrict__ s, int kt,
                                           int lane) {
  int K0 = 32 * kt + ((lane & 16) ? 16 : 0);
  v8h s0 = *(const v8h*)(s + K0);
  v8h s1 = *(const v8h*)(s + K0 + 8);
  return __builtin_shufflevector(s0, s1, 0, 1, 2, 3, 4, 5, 6, 7,
                                 8, 9, 10, 11, 12, 13, 14, 15);
}

__global__ void __launch_bounds__(32)
spike_kernel(const float* __restrict__ w69, const float* __restrict__ start,
             float* __restrict__ out, int steps) {
  const int lane = threadIdx.x;
  const int l15  = lane & 15;
  __shared__ __align__(16) _Float16 s_new[128];

  // Resident B tiles (kt = K/32 tile, nt = N/16 tile)
  v16h B00 = load_btile(w69, lane, 0, 0), B01 = load_btile(w69, lane, 0, 1),
       B02 = load_btile(w69, lane, 0, 2), B03 = load_btile(w69, lane, 0, 3),
       B04 = load_btile(w69, lane, 0, 4);
  v16h B10 = load_btile(w69, lane, 1, 0), B11 = load_btile(w69, lane, 1, 1),
       B12 = load_btile(w69, lane, 1, 2), B13 = load_btile(w69, lane, 1, 3),
       B14 = load_btile(w69, lane, 1, 4);
  v16h B20 = load_btile(w69, lane, 2, 0), B21 = load_btile(w69, lane, 2, 1),
       B22 = load_btile(w69, lane, 2, 2), B23 = load_btile(w69, lane, 2, 3),
       B24 = load_btile(w69, lane, 2, 4);

  // prev state: lanes 0-15 own indices nt*16 + lane (lanes 16-31 mirror)
  float p0, p1, p2, p3, p4;
  {
    const float* s0 = start + 199 * NN;
    int i4 = 64 + l15;
    p0 = s0[l15]; p1 = s0[16 + l15]; p2 = s0[32 + l15]; p3 = s0[48 + l15];
    p4 = (i4 < NN) ? s0[i4] : 0.f;
  }
  // zero-pad + fill LDS state (K padding 69..127 stays 0 forever)
  s_new[lane] = (_Float16)0.f;       s_new[lane + 32] = (_Float16)0.f;
  s_new[lane + 64] = (_Float16)0.f;  s_new[lane + 96] = (_Float16)0.f;
  __syncthreads();
  if (lane < 16) {
    s_new[l15]      = (_Float16)p0;
    s_new[16 + l15] = (_Float16)p1;
    s_new[32 + l15] = (_Float16)p2;
    s_new[48 + l15] = (_Float16)p3;
    if (64 + l15 < NN) s_new[64 + l15] = (_Float16)p4;
  }
  __syncthreads();

  const v8f zero = {0.f, 0.f, 0.f, 0.f, 0.f, 0.f, 0.f, 0.f};
  for (int t = 0; t < steps; ++t) {
    v16h a0 = load_atile(s_new, 0, lane);
    v16h a1 = load_atile(s_new, 1, lane);
    v16h a2 = load_atile(s_new, 2, lane);

    v8f c0 = zero, c1 = zero, c2 = zero, c3 = zero, c4 = zero;
    c0 = wmma_f16(a0, B00, c0); c1 = wmma_f16(a0, B01, c1);
    c2 = wmma_f16(a0, B02, c2); c3 = wmma_f16(a0, B03, c3);
    c4 = wmma_f16(a0, B04, c4);
    c0 = wmma_f16(a1, B10, c0); c1 = wmma_f16(a1, B11, c1);
    c2 = wmma_f16(a1, B12, c2); c3 = wmma_f16(a1, B13, c3);
    c4 = wmma_f16(a1, B14, c4);
    c0 = wmma_f16(a2, B20, c0); c1 = wmma_f16(a2, B21, c1);
    c2 = wmma_f16(a2, B22, c2); c3 = wmma_f16(a2, B23, c3);
    c4 = wmma_f16(a2, B24, c4);

    // D row M=0 lives in v8f element 0, lanes 0-15 (branch-free tanh)
    float n0 = tanh_fast(c0[0]) - p0;
    float n1 = tanh_fast(c1[0]) - p1;
    float n2 = tanh_fast(c2[0]) - p2;
    float n3 = tanh_fast(c3[0]) - p3;
    float n4 = tanh_fast(c4[0]) - p4;

    wave_lds_fence();  // all lanes' A loads done (HW: LDS is in-order per wave)
    if (lane < 16) {
      float* orow = out + (size_t)t * NN;
      s_new[l15]      = (_Float16)n0;  orow[l15]      = n0;
      s_new[16 + l15] = (_Float16)n1;  orow[16 + l15] = n1;
      s_new[32 + l15] = (_Float16)n2;  orow[32 + l15] = n2;
      s_new[48 + l15] = (_Float16)n3;  orow[48 + l15] = n3;
      if (64 + l15 < NN) { s_new[64 + l15] = (_Float16)n4; orow[64 + l15] = n4; }
    }
    p0 = n0; p1 = n1; p2 = n2; p3 = n3; p4 = n4;
    wave_lds_fence();  // stores ordered before next step's A loads
  }
}

// ---------------------------------------------------------------------------
extern "C" void kernel_launch(void* const* d_in, const int* in_sizes, int n_in,
                              void* d_out, int out_size, void* d_ws, size_t ws_size,
                              hipStream_t stream) {
  const float* x     = (const float*)d_in[0];
  const float* start = (const float*)d_in[1];
  const float* W_in  = (const float*)d_in[2];
  const float* b_in  = (const float*)d_in[3];
  const float* w1    = (const float*)d_in[4];
  const float* w2    = (const float*)d_in[5];
  const float* w3    = (const float*)d_in[6];
  const float* w4    = (const float*)d_in[7];
  const float* w5    = (const float*)d_in[8];
  const float* g1 = (const float*)d_in[9],  *be1 = (const float*)d_in[10];
  const float* g2 = (const float*)d_in[11], *be2 = (const float*)d_in[12];
  const float* g3 = (const float*)d_in[13], *be3 = (const float*)d_in[14];
  const float* g4 = (const float*)d_in[15], *be4 = (const float*)d_in[16];
  const float* W_d2  = (const float*)d_in[17];
  const float* b_d2  = (const float*)d_in[18];
  float* out = (float*)d_out;

  // workspace layout (floats)
  float* ws   = (float*)d_ws;
  float* h1   = ws;                  // 12800 (also reused for 6400 conv5 out)
  float* bufA = h1 + 12800;          // 409600
  float* bufB = bufA + 409600;       // 409600
  float* mean = bufB + 409600;       // 512
  float* rstd = mean + 512;          // 512
  float* w69  = rstd + 512;          // 4761

  const int TB = 256;
  // 1) h1 = W_in @ x + b_in  (12800 x 2048)
  gemv_kernel<<<(12800 * WAVE + TB - 1) / TB, TB, 0, stream>>>(W_in, x, b_in, h1,
                                                               12800, 2048);
  // 2) stage 1: (512,5,5) -> up -> conv w1 -> (512,10,10)
  upconv_kernel<<<(512 * 100 + TB - 1) / TB, TB, 0, stream>>>(h1, w1, bufA, 512, 512, 10);
  bnstats_kernel<<<512, TB, 0, stream>>>(bufA, mean, rstd, 100);
  bnrelu_kernel<<<(512 * 100 + TB - 1) / TB, TB, 0, stream>>>(bufA, mean, rstd, g1, be1,
                                                              100, 512 * 100);
  // 3) stage 2: -> (256,20,20)
  upconv_kernel<<<(256 * 400 + TB - 1) / TB, TB, 0, stream>>>(bufA, w2, bufB, 512, 256, 20);
  bnstats_kernel<<<256, TB, 0, stream>>>(bufB, mean, rstd, 400);
  bnrelu_kernel<<<(256 * 400 + TB - 1) / TB, TB, 0, stream>>>(bufB, mean, rstd, g2, be2,
                                                              400, 256 * 400);
  // 4) stage 3: -> (128,40,40)
  upconv_kernel<<<(128 * 1600 + TB - 1) / TB, TB, 0, stream>>>(bufB, w3, bufA, 256, 128, 40);
  bnstats_kernel<<<128, TB, 0, stream>>>(bufA, mean, rstd, 1600);
  bnrelu_kernel<<<(128 * 1600 + TB - 1) / TB, TB, 0, stream>>>(bufA, mean, rstd, g3, be3,
                                                               1600, 128 * 1600);
  // 5) stage 4: -> (64,80,80)
  upconv_kernel<<<(64 * 6400 + TB - 1) / TB, TB, 0, stream>>>(bufA, w4, bufB, 128, 64, 80);
  bnstats_kernel<<<64, TB, 0, stream>>>(bufB, mean, rstd, 6400);
  bnrelu_kernel<<<(64 * 6400 + TB - 1) / TB, TB, 0, stream>>>(bufB, mean, rstd, g4, be4,
                                                              6400, 64 * 6400);
  // 6) final conv + tanh -> h1[0..6400)
  conv_tanh_kernel<<<(6400 + TB - 1) / TB, TB, 0, stream>>>(bufB, w5, h1, 64, 80);
  // 7) w69 = W_d2 @ h + b_d2  (4761 x 6400)
  gemv_kernel<<<(4761 * WAVE + TB - 1) / TB, TB, 0, stream>>>(W_d2, h1, b_d2, w69,
                                                              4761, 6400);
  // 8) sequential spiking recurrence (single wave, WMMA matvec per step)
  int steps = out_size / NN;   // 99800
  spike_kernel<<<1, 32, 0, stream>>>(w69, start, out, steps);

  (void)in_sizes; (void)n_in; (void)ws_size;
}
